// STSN_85349590106470
// MI455X (gfx1250) — matrix operand order repaired and checked by reading, not compiled, not run on hardware
//
#include <hip/hip_runtime.h>

typedef __attribute__((ext_vector_type(16))) _Float16 v16h;
typedef __attribute__((ext_vector_type(8)))  _Float16 v8h;
typedef __attribute__((ext_vector_type(8)))  float    v8f;

#define HW     4096     // 64*64
#define BATCH  4
#define NEG_SLOPE 0.01f
#define GN_EPS 1e-5f

// ---------------------------------------------------------------------------
// features = concat([support, ref], axis=1) : (4,512,64,64)
// ---------------------------------------------------------------------------
__global__ __launch_bounds__(256) void concat_kernel(
    const float* __restrict__ a, const float* __restrict__ c, float* __restrict__ out)
{
    int idx = blockIdx.x * 256 + threadIdx.x;           // < 4*512*4096
    int b   = idx >> 21;                                // / (512*4096)
    int r   = idx & ((1 << 21) - 1);
    int ch  = r >> 12;
    int n   = r & (HW - 1);
    float v;
    if (ch < 256) v = a[(((size_t)b * 256 + ch) << 12) + n];
    else          v = c[(((size_t)b * 256 + (ch - 256)) << 12) + n];
    out[idx] = v;
}

// ---------------------------------------------------------------------------
// Repack conv weights w[o][ci][t] (t = ky*3+kx) -> f16 Wk[o][t*Cin + ci]
// so each K-block of 32 is a single tap over 32 consecutive input channels.
// ---------------------------------------------------------------------------
__global__ __launch_bounds__(256) void pack_weights_kernel(
    const float* __restrict__ w, _Float16* __restrict__ wk, int Cin)
{
    int idx = blockIdx.x * 256 + threadIdx.x;           // < Cout*Cin*9
    int o   = idx / (Cin * 9);
    int rem = idx - o * (Cin * 9);
    int ci  = rem / 9;
    int t   = rem - ci * 9;
    wk[(size_t)o * (9 * Cin) + t * Cin + ci] = (_Float16)w[idx];
}

// ---------------------------------------------------------------------------
// Offset conv: 3x3, Cin=512 -> 18 channels.  Block = 256 consecutive pixels
// for a fixed (b, co); weights are wave-uniform (scalar loads), x is L2-hot.
// ---------------------------------------------------------------------------
__global__ __launch_bounds__(256) void offset_conv_kernel(
    const float* __restrict__ x, const float* __restrict__ w,
    const float* __restrict__ bias, float* __restrict__ out)
{
    const int n  = blockIdx.x * 256 + threadIdx.x;      // 0..4095
    const int co = blockIdx.y;                          // 0..17
    const int b  = blockIdx.z;
    const int h  = n >> 6, wp = n & 63;
    float acc = bias[co];
    const float* wrow = w + (size_t)co * 512 * 9;
    for (int ci = 0; ci < 512; ++ci) {
        const float* xp = x + (((size_t)b * 512 + ci) << 12);
        const float* wc = wrow + ci * 9;
        #pragma unroll
        for (int ky = 0; ky < 3; ++ky) {
            int yy = h + ky - 1;
            if (yy < 0 || yy > 63) continue;
            #pragma unroll
            for (int kx = 0; kx < 3; ++kx) {
                int xx = wp + kx - 1;
                if (xx < 0 || xx > 63) continue;
                acc += xp[(yy << 6) + xx] * wc[ky * 3 + kx];
            }
        }
    }
    out[(((size_t)b * 18 + co) << 12) + n] = acc;
}

// ---------------------------------------------------------------------------
// Fused deformable conv as WMMA GEMM.
//   D[Cout, HW] = Wk[Cout, 9*Cin] x S[9*Cin, HW]
// Block: 256 threads = 8 waves; tile = 256 Cout rows x 32 pixels.
// Each wave owns rows [m0,m0+16) and [m0+128,m0+144) -> 4 accumulators.
// Per K-block (one tap, 32 cin): bilinear-sample a 32x32 f16 tile into a
// double-buffered LDS tile (stride 40 halfs: 16B-aligned, conflict-free
// column reads), then 4x v_wmma_f32_16x16x32_f16 per wave.  One barrier
// per K-block: iteration kb reads buf p while sampling kb+1 into buf p^1.
// ---------------------------------------------------------------------------
template <int CIN>
__global__ __launch_bounds__(256) void deform_wmma_kernel(
    const float* __restrict__ x,      // (B, CIN, 64, 64)
    const float* __restrict__ offs,   // (B, 18, 64, 64)
    const _Float16* __restrict__ wk,  // (Cout, 9*CIN) packed f16
    float* __restrict__ Y,            // (B, Cout, 64, 64)
    int Cout)
{
    const int NKB = 9 * CIN / 32;                // K-blocks (144 / 72)
    const int CBM = CIN / 32 - 1;                // cin-block mask (pow2-1)
    const int tid    = threadIdx.x;
    const int lane   = tid & 31;
    const int wave   = tid >> 5;                 // 0..7
    const int laneM  = lane & 15;
    const int laneHi = lane >> 4;
    const int b      = blockIdx.z;
    const int n0     = blockIdx.x * 32;          // pixel base (half image row)
    const int m0     = blockIdx.y * 256 + wave * 16;
    const int KW     = 9 * CIN;

    __shared__ __attribute__((aligned(32))) _Float16 sS[2][32 * 40];
    __shared__ int   sY0[9 * 32];
    __shared__ int   sX0[9 * 32];
    __shared__ float sWy[9 * 32];
    __shared__ float sWx[9 * 32];

    for (int i = tid; i < 9 * 32; i += 256) {
        int t  = i >> 5, px = i & 31;
        int n  = n0 + px;
        int h  = n >> 6, wp = n & 63;
        float dy = offs[(((size_t)b * 18 + 2 * t)     << 12) + n];
        float dx = offs[(((size_t)b * 18 + 2 * t + 1) << 12) + n];
        float sy = (float)h + (float)(t / 3 - 1) + dy;
        float sx = (float)wp + (float)(t % 3 - 1) + dx;
        float fy = floorf(sy), fx = floorf(sx);
        sY0[i] = (int)fy;  sX0[i] = (int)fx;
        sWy[i] = sy - fy;  sWx[i] = sx - fx;
    }
    __syncthreads();

    const int px = tid & 31;                     // sampling pixel for this thread
    const float* xbat = x + ((size_t)b * CIN << 12);

    // Sample K-block kb (tap t, 32 cin) into LDS buffer buf.
    auto sample_tile = [&](int kb, _Float16* buf) {
        int t  = kb / (CBM + 1);
        int cb = kb & CBM;
        const int   mi = t * 32 + px;
        const int   y0 = sY0[mi], x0 = sX0[mi];
        const float wy = sWy[mi], wx = sWx[mi];
        const int   y1 = y0 + 1,  x1 = x0 + 1;
        const bool  vy0 = (unsigned)y0 < 64u, vy1 = (unsigned)y1 < 64u;
        const bool  vx0 = (unsigned)x0 < 64u, vx1 = (unsigned)x1 < 64u;
        const int y0c = y0 < 0 ? 0 : (y0 > 63 ? 63 : y0);
        const int y1c = y1 < 0 ? 0 : (y1 > 63 ? 63 : y1);
        const int x0c = x0 < 0 ? 0 : (x0 > 63 ? 63 : x0);
        const int x1c = x1 < 0 ? 0 : (x1 > 63 ? 63 : x1);
        const float m00 = (vy0 && vx0) ? (1.f - wy) * (1.f - wx) : 0.f;
        const float m01 = (vy0 && vx1) ? (1.f - wy) * wx         : 0.f;
        const float m10 = (vy1 && vx0) ? wy * (1.f - wx)         : 0.f;
        const float m11 = (vy1 && vx1) ? wy * wx                 : 0.f;
        const int o00 = (y0c << 6) + x0c, o01 = (y0c << 6) + x1c;
        const int o10 = (y1c << 6) + x0c, o11 = (y1c << 6) + x1c;
        #pragma unroll
        for (int j = 0; j < 4; ++j) {
            int cl = wave + (j << 3);                          // 0..31
            const float* xc = xbat + ((size_t)(cb * 32 + cl) << 12);
            float v = m00 * xc[o00] + m01 * xc[o01]
                    + m10 * xc[o10] + m11 * xc[o11];
            buf[px * 40 + cl] = (_Float16)v;
        }
    };

    v8f acc00 = {}, acc01 = {}, acc10 = {}, acc11 = {};
    const _Float16* wrow0 = wk + (size_t)(m0 + laneM) * KW + laneHi * 8;
    const _Float16* wrow1 = wrow0 + ((size_t)128 * KW);

    sample_tile(0, sS[0]);
    __syncthreads();

    int p = 0;
    for (int kb = 0; kb < NKB; ++kb) {
        // A fragments: rows m0.. and m0+128.. (L2-resident packed weights)
        const _Float16* wr0 = wrow0 + kb * 32;
        const _Float16* wr1 = wrow1 + kb * 32;
        __builtin_prefetch(wr0 + 32, 0, 1);
        __builtin_prefetch(wr1 + 32, 0, 1);
        union { v16h v; v8h h[2]; } A0, A1, B0, B1;
        A0.h[0] = *(const v8h*)(wr0);  A0.h[1] = *(const v8h*)(wr0 + 16);
        A1.h[0] = *(const v8h*)(wr1);  A1.h[1] = *(const v8h*)(wr1 + 16);

        // B fragments from current LDS buffer: col = laneM (+16),
        // K = 8*half + {0..7} and 16 + 8*half + {0..7} (matches A).
        const _Float16* bp0 = &sS[p][laneM * 40 + laneHi * 8];
        const _Float16* bp1 = &sS[p][(laneM + 16) * 40 + laneHi * 8];
        B0.h[0] = *(const v8h*)bp0;  B0.h[1] = *(const v8h*)(bp0 + 16);
        B1.h[0] = *(const v8h*)bp1;  B1.h[1] = *(const v8h*)(bp1 + 16);

        // Overlap: sample next K-block into the other buffer.
        if (kb + 1 < NKB) sample_tile(kb + 1, sS[p ^ 1]);

        acc00 = __builtin_amdgcn_wmma_f32_16x16x32_f16(
            false, A0.v, false, B0.v, (short)0, acc00, false, false);
        acc01 = __builtin_amdgcn_wmma_f32_16x16x32_f16(
            false, A0.v, false, B1.v, (short)0, acc01, false, false);
        acc10 = __builtin_amdgcn_wmma_f32_16x16x32_f16(
            false, A1.v, false, B0.v, (short)0, acc10, false, false);
        acc11 = __builtin_amdgcn_wmma_f32_16x16x32_f16(
            false, A1.v, false, B1.v, (short)0, acc11, false, false);

        __syncthreads();
        p ^= 1;
    }

    // C/D layout: VGPR r, lane l -> M = r + (l>=16 ? 8 : 0), N = l%16.
    float* yb = Y + (((size_t)b * Cout) << 12) + n0;
    #pragma unroll
    for (int r = 0; r < 8; ++r) {
        int mA = m0 + r + laneHi * 8;
        int mB = mA + 128;
        yb[((size_t)mA << 12) + laneM]      = acc00[r];
        yb[((size_t)mA << 12) + 16 + laneM] = acc01[r];
        yb[((size_t)mB << 12) + laneM]      = acc10[r];
        yb[((size_t)mB << 12) + 16 + laneM] = acc11[r];
    }
}

// ---------------------------------------------------------------------------
// GroupNorm stats: one block per (b, group); tree-reduce -> mean, rstd.
// ---------------------------------------------------------------------------
__global__ __launch_bounds__(256) void gn_stats_kernel(
    const float* __restrict__ Y, float* __restrict__ stats, int C)
{
    const int bg = blockIdx.x;                   // b*32 + g
    const int b  = bg >> 5, g = bg & 31;
    const int cpg = C >> 5;
    const size_t base = ((size_t)b * C + (size_t)g * cpg) << 12;
    const int n = cpg << 12;

    float s = 0.f, ss = 0.f;
    for (int i = threadIdx.x; i < n; i += 256) {
        float v = Y[base + i];
        s += v; ss += v * v;
    }
    __shared__ float rs[256], rq[256];
    rs[threadIdx.x] = s; rq[threadIdx.x] = ss;
    __syncthreads();
    for (int st = 128; st > 0; st >>= 1) {
        if (threadIdx.x < st) {
            rs[threadIdx.x] += rs[threadIdx.x + st];
            rq[threadIdx.x] += rq[threadIdx.x + st];
        }
        __syncthreads();
    }
    if (threadIdx.x == 0) {
        float inv  = 1.f / (float)n;
        float mean = rs[0] * inv;
        float var  = rq[0] * inv - mean * mean;
        stats[bg * 2 + 0] = mean;
        stats[bg * 2 + 1] = rsqrtf(var + GN_EPS);
    }
}

// ---------------------------------------------------------------------------
// GroupNorm apply + affine + LeakyReLU.
// ---------------------------------------------------------------------------
__global__ __launch_bounds__(256) void gn_apply_kernel(
    const float* __restrict__ Y, const float* __restrict__ stats,
    const float* __restrict__ gamma, const float* __restrict__ beta,
    float* __restrict__ out, int cLog2)
{
    const int idx = blockIdx.x * 256 + threadIdx.x;      // < B*C*4096
    const int c = (idx >> 12) & ((1 << cLog2) - 1);
    const int b = idx >> (12 + cLog2);
    const int g = c >> (cLog2 - 5);
    const float mean = stats[(b * 32 + g) * 2 + 0];
    const float rstd = stats[(b * 32 + g) * 2 + 1];
    float v = (Y[idx] - mean) * rstd * gamma[c] + beta[c];
    out[idx] = v >= 0.f ? v : NEG_SLOPE * v;
}

// ---------------------------------------------------------------------------
// Host orchestration.
// ---------------------------------------------------------------------------
extern "C" void kernel_launch(void* const* d_in, const int* in_sizes, int n_in,
                              void* d_out, int out_size, void* d_ws, size_t ws_size,
                              hipStream_t stream)
{
    const float* support = (const float*)d_in[0];
    const float* refin   = (const float*)d_in[1];
    const float* W_OFF[4] = {(const float*)d_in[2],  (const float*)d_in[7],
                             (const float*)d_in[12], (const float*)d_in[17]};
    const float* B_OFF[4] = {(const float*)d_in[3],  (const float*)d_in[8],
                             (const float*)d_in[13], (const float*)d_in[18]};
    const float* W_DC[4]  = {(const float*)d_in[4],  (const float*)d_in[9],
                             (const float*)d_in[14], (const float*)d_in[19]};
    const float* GAM[4]   = {(const float*)d_in[5],  (const float*)d_in[10],
                             (const float*)d_in[15], (const float*)d_in[20]};
    const float* BET[4]   = {(const float*)d_in[6],  (const float*)d_in[11],
                             (const float*)d_in[16], (const float*)d_in[21]};

    // Workspace layout (~174 MB total, float units)
    const size_t FEAT = (size_t)BATCH * 512 * HW;        // 8,388,608 floats
    float* ws       = (float*)d_ws;
    float* features = ws;
    float* f1       = features + FEAT;
    float* f2       = f1 + FEAT;
    float* f3       = f2 + FEAT;
    float* Yraw     = f3 + FEAT;
    float* offs     = Yraw + FEAT;                       // B*18*HW = 294,912
    float* stats    = offs + (size_t)BATCH * 18 * HW;    // B*32*2  = 256
    _Float16* wkf16 = (_Float16*)(stats + 256);          // up to 512*4608 halfs

    // features = concat(support, ref)
    concat_kernel<<<(int)(2 * (size_t)BATCH * 256 * HW / 256), 256, 0, stream>>>(
        support, refin, features);

    struct StageCfg {
        const float* x_off;  // offset-conv input (always 512 ch)
        const float* x_dc;   // deform-conv sampled input
        float*       out;
        int Cin, Cout, cLog2;
    };
    StageCfg st[4] = {
        { features, features, f1,            512, 512, 9 },
        { f1,       f1,       f2,            512, 512, 9 },
        { f2,       f2,       f3,            512, 512, 9 },
        { f3,       support,  (float*)d_out, 256, 256, 8 },
    };

    for (int s = 0; s < 4; ++s) {
        const int Cin = st[s].Cin, Cout = st[s].Cout;

        pack_weights_kernel<<<(int)((size_t)Cout * Cin * 9 / 256), 256, 0, stream>>>(
            W_DC[s], wkf16, Cin);

        offset_conv_kernel<<<dim3(HW / 256, 18, BATCH), 256, 0, stream>>>(
            st[s].x_off, W_OFF[s], B_OFF[s], offs);

        dim3 dg(HW / 32, Cout / 256, BATCH);
        if (Cin == 512)
            deform_wmma_kernel<512><<<dg, 256, 0, stream>>>(
                st[s].x_dc, offs, wkf16, Yraw, Cout);
        else
            deform_wmma_kernel<256><<<dg, 256, 0, stream>>>(
                st[s].x_dc, offs, wkf16, Yraw, Cout);

        gn_stats_kernel<<<BATCH * 32, 256, 0, stream>>>(Yraw, stats, Cout);

        gn_apply_kernel<<<(int)((size_t)BATCH * Cout * HW / 256), 256, 0, stream>>>(
            Yraw, stats, GAM[s], BET[s], st[s].out, st[s].cLog2);
    }
}
